// CausalSelfAttention_5368709120652
// MI455X (gfx1250) — compile-verified, matrix operand order checked by reading, not run on hardware
//
#include <hip/hip_runtime.h>
#include <hip/hip_bf16.h>

typedef __attribute__((ext_vector_type(16))) __bf16 v16bf;
typedef __attribute__((ext_vector_type(4)))  __bf16 v4bf;
typedef __attribute__((ext_vector_type(8)))  float  v8f;

#define TSEQ   2048
#define BATCH  4
#define CEMB   1024
#define C3     3072
#define NHEAD  16
#define HD     64
#define MTOT   (BATCH*TSEQ)   // 8192

// ---------------------------------------------------------------------------
// WMMA helper: D = A(16x32 bf16) * B(32x16 bf16) + C(16x16 f32)
// ---------------------------------------------------------------------------
static __device__ __forceinline__ v8f wmma_bf16(v16bf a, v16bf b, v8f c) {
    return __builtin_amdgcn_wmma_f32_16x16x32_bf16(
        /*neg_a=*/false, a, /*neg_b=*/false, b,
        /*c_mod=*/(short)0, c, /*reuse_a=*/false, /*reuse_b=*/false);
}

// Async global -> LDS DMA (CDNA5). Per-lane: 16 bytes from gaddr to LDS offset.
// Tracked by ASYNCcnt; completion enforced with s_wait_asynccnt + barrier.
static __device__ __forceinline__ void async_ld_b128(unsigned int lds_off,
                                                     unsigned long long gaddr) {
    asm volatile("global_load_async_to_lds_b128 %0, %1, off"
                 :: "v"(lds_off), "v"(gaddr) : "memory");
}
static __device__ __forceinline__ void wait_async_le4() {
    asm volatile("s_wait_asynccnt 0x4" ::: "memory");
}
static __device__ __forceinline__ void wait_async_le0() {
    asm volatile("s_wait_asynccnt 0x0" ::: "memory");
}

// Low 32 bits of a generic (shared-aperture) address == LDS byte offset.
static __device__ __forceinline__ unsigned int lds_off(const void* p) {
    return (unsigned int)(unsigned long long)(uintptr_t)p;
}

// Load a 16x32 bf16 fragment (A layout; B layout is symmetric over N) from a
// row-major (K contiguous) tile with leading dimension `ld` (in elements).
// ISA layout: lanes 0-15 hold row=lane, K in {0..7, 16..23};
//             lanes 16-31 hold row=lane-16, K in {8..15, 24..31}.
static __device__ __forceinline__ v16bf frag_ld(const __bf16* base, int ld) {
    const int lane = threadIdx.x & 31;
    const int row  = lane & 15;
    const int kb   = (lane >> 4) << 3;          // 0 or 8
    const __bf16* p = base + row * ld + kb;
    v16bf f;
    ((float4*)&f)[0] = *(const float4*)(p);      // K = kb .. kb+7
    ((float4*)&f)[1] = *(const float4*)(p + 16); // K = 16+kb .. 16+kb+7
    return f;
}

// ---------------------------------------------------------------------------
// f32 -> bf16 conversion (vectorized x4)
// ---------------------------------------------------------------------------
__global__ void cvt_f32_bf16(const float* __restrict__ s,
                             __bf16* __restrict__ d, int n4) {
    int i = blockIdx.x * blockDim.x + threadIdx.x;
    if (i >= n4) return;
    float4 v = ((const float4*)s)[i];
    v4bf o;
    o[0] = (__bf16)v.x; o[1] = (__bf16)v.y;
    o[2] = (__bf16)v.z; o[3] = (__bf16)v.w;
    ((v4bf*)d)[i] = o;
}

// ---------------------------------------------------------------------------
// GEMM: C[M,N] = A[M,K](bf16,row-major) * Bm[N,K]^T (bf16,row-major) + bias[N]
// Block tile 128x128, BK=32, double-buffered LDS staging via
// GLOBAL_LOAD_ASYNC_TO_LDS_B128 (ASYNCcnt). 256 threads = 8 waves (2M x 4N);
// each wave owns a 64x32 tile = 4x2 WMMA accumulators.
// ---------------------------------------------------------------------------
__global__ __launch_bounds__(256) void gemm_bf16_abT(
    const __bf16* __restrict__ A, const __bf16* __restrict__ Bm,
    const float* __restrict__ bias, float* __restrict__ C,
    int M, int N, int K)
{
    __shared__ __align__(16) __bf16 As[2][128 * 32];
    __shared__ __align__(16) __bf16 Bs[2][128 * 32];

    const int tid  = threadIdx.x;
    const int w    = tid >> 5;
    const int lane = tid & 31;
    const int wm   = w & 1;        // 0..1
    const int wn   = w >> 1;       // 0..3
    const int m0   = blockIdx.y * 128;
    const int n0   = blockIdx.x * 128;

    v8f acc[4][2];
    #pragma unroll
    for (int sm = 0; sm < 4; ++sm)
        #pragma unroll
        for (int sn = 0; sn < 2; ++sn)
            #pragma unroll
            for (int j = 0; j < 8; ++j) acc[sm][sn][j] = 0.0f;

    // Issue one stage of async tile loads into buffer `buf` for k-offset k0.
    // 512 16B chunks per operand; 2 A-chunks + 2 B-chunks per thread
    // => 4 outstanding ASYNCcnt entries per wave per stage.
    auto issue = [&](int buf, int k0) {
        #pragma unroll
        for (int c = tid; c < 512; c += 256) {
            int row = c >> 2, col = c & 3;     // col indexes 8-elem chunks
            async_ld_b128(
                lds_off(&As[buf][row * 32 + col * 8]),
                (unsigned long long)(uintptr_t)(A + (size_t)(m0 + row) * K + k0 + col * 8));
            async_ld_b128(
                lds_off(&Bs[buf][row * 32 + col * 8]),
                (unsigned long long)(uintptr_t)(Bm + (size_t)(n0 + row) * K + k0 + col * 8));
        }
    };

    const int NS = K >> 5;         // k-steps of 32
    issue(0, 0);
    for (int i = 0; i < NS; ++i) {
        const int cur = i & 1;
        if (i + 1 < NS) {
            issue(cur ^ 1, (i + 1) << 5);  // prefetch next stage
            wait_async_le4();              // current stage done; next in flight
        } else {
            wait_async_le0();
        }
        __syncthreads();                   // all waves' DMA visible in LDS

        v16bf bfr[2];
        bfr[0] = frag_ld(&Bs[cur][(wn * 32 +  0) * 32], 32);
        bfr[1] = frag_ld(&Bs[cur][(wn * 32 + 16) * 32], 32);
        #pragma unroll
        for (int sm = 0; sm < 4; ++sm) {
            v16bf afr = frag_ld(&As[cur][(wm * 64 + sm * 16) * 32], 32);
            acc[sm][0] = wmma_bf16(afr, bfr[0], acc[sm][0]);
            acc[sm][1] = wmma_bf16(afr, bfr[1], acc[sm][1]);
        }
        __syncthreads();                   // done reading `cur` before reuse
    }

    // Epilogue: C/D layout -> row = sub_base + (lane>=16?8:0) + j, col = lane&15
    const int laneh = lane >> 4, lc = lane & 15;
    #pragma unroll
    for (int sn = 0; sn < 2; ++sn) {
        int n = n0 + wn * 32 + sn * 16 + lc;
        float bv = bias[n];
        #pragma unroll
        for (int sm = 0; sm < 4; ++sm) {
            int mb = m0 + wm * 64 + sm * 16 + laneh * 8;
            #pragma unroll
            for (int j = 0; j < 8; ++j)
                C[(size_t)(mb + j) * N + n] = acc[sm][sn][j] + bv;
        }
    }
}

// ---------------------------------------------------------------------------
// Flash attention: grid (T/64, H, B), 128 threads (4 waves).
// Wave w owns query rows [qt*64 + w*16, +16). Keys processed in blocks of 64.
// S and P never touch HBM. y written as bf16 for the output projection.
// (Staging stays synchronous: it must convert f32 -> bf16 in flight.)
// ---------------------------------------------------------------------------
__global__ __launch_bounds__(128) void attn_flash(
    const float* __restrict__ qkv,    // [B,T,3C] f32
    __bf16* __restrict__ yb)          // [B,T,C] bf16
{
    __shared__ __align__(16) __bf16 Qs[64 * 64];       // [qrow][hd], pre-scaled
    __shared__ __align__(16) __bf16 Kt[64 * 64];       // [key][hd]
    __shared__ __align__(16) __bf16 Vt[64 * 64];       // [hd][key] (transposed)
    __shared__ __align__(16) __bf16 Pb[4][16 * 64];    // per-wave P staging

    const int qt = blockIdx.x, h = blockIdx.y, b = blockIdx.z;
    const int tid = threadIdx.x;
    const int w = tid >> 5, lane = tid & 31;
    const int laneh = lane >> 4, lc = lane & 15;

    const size_t rowQ = (size_t)b * TSEQ;              // row base in [B*T]
    const size_t hoff = (size_t)h * HD;

    // Stage Q tile (scaled by 1/sqrt(hd) = 0.125 so S comes out pre-scaled)
    for (int c = tid; c < 1024; c += 128) {
        int row = c >> 4, f4 = c & 15;
        const float* p = qkv + (rowQ + qt * 64 + row) * C3 + hoff + f4 * 4;
        float4 v = *(const float4*)p;
        int o = row * 64 + f4 * 4;
        Qs[o + 0] = (__bf16)(v.x * 0.125f);
        Qs[o + 1] = (__bf16)(v.y * 0.125f);
        Qs[o + 2] = (__bf16)(v.z * 0.125f);
        Qs[o + 3] = (__bf16)(v.w * 0.125f);
    }

    v8f O[4];
    float mrow[8], lrow[8];
    #pragma unroll
    for (int t = 0; t < 4; ++t)
        #pragma unroll
        for (int j = 0; j < 8; ++j) O[t][j] = 0.0f;
    #pragma unroll
    for (int j = 0; j < 8; ++j) { mrow[j] = -1e30f; lrow[j] = 0.0f; }

    for (int kb = 0; kb <= qt; ++kb) {
        __syncthreads();   // prior iter done reading Kt/Vt (and Q staged)
        // Stage K (row-major) and V (transposed) for this key block
        for (int c = tid; c < 1024; c += 128) {
            int row = c >> 4, f4 = c & 15;
            const float* p = qkv + (rowQ + kb * 64 + row) * C3 + CEMB + hoff + f4 * 4;
            float4 kv = *(const float4*)p;
            float4 vv = *(const float4*)(p + CEMB);   // V is +C after K
            int o = row * 64 + f4 * 4;
            Kt[o + 0] = (__bf16)kv.x; Kt[o + 1] = (__bf16)kv.y;
            Kt[o + 2] = (__bf16)kv.z; Kt[o + 3] = (__bf16)kv.w;
            int hd0 = f4 * 4;
            Vt[(hd0 + 0) * 64 + row] = (__bf16)vv.x;
            Vt[(hd0 + 1) * 64 + row] = (__bf16)vv.y;
            Vt[(hd0 + 2) * 64 + row] = (__bf16)vv.z;
            Vt[(hd0 + 3) * 64 + row] = (__bf16)vv.w;
        }
        __syncthreads();

        // S = Q * K^T : 16 query rows x 64 keys, K-dim = hd = 64 (2 chunks)
        v8f S[4];
        #pragma unroll
        for (int t = 0; t < 4; ++t)
            #pragma unroll
            for (int j = 0; j < 8; ++j) S[t][j] = 0.0f;
        #pragma unroll
        for (int kc = 0; kc < 64; kc += 32) {
            v16bf aq = frag_ld(Qs + (w * 16) * 64 + kc, 64);
            #pragma unroll
            for (int t = 0; t < 4; ++t) {
                v16bf bk = frag_ld(Kt + (t * 16) * 64 + kc, 64);
                S[t] = wmma_bf16(aq, bk, S[t]);
            }
        }

        // Causal mask + online softmax. Row r data sits in a 16-lane half.
        const int qrow0 = qt * 64 + w * 16 + laneh * 8;
        const int col0  = kb * 64 + lc;
        #pragma unroll
        for (int j = 0; j < 8; ++j) {
            float mx = mrow[j];
            #pragma unroll
            for (int t = 0; t < 4; ++t) {
                float s = S[t][j];
                s = (col0 + t * 16 > qrow0 + j) ? -1e30f : s;
                S[t][j] = s;
                mx = fmaxf(mx, s);
            }
            #pragma unroll
            for (int msk = 1; msk < 16; msk <<= 1)
                mx = fmaxf(mx, __shfl_xor(mx, msk));
            float alpha = __expf(mrow[j] - mx);
            float rs = 0.0f;
            #pragma unroll
            for (int t = 0; t < 4; ++t) {
                float p = __expf(S[t][j] - mx);
                Pb[w][(laneh * 8 + j) * 64 + t * 16 + lc] = (__bf16)p;
                rs += p;
            }
            #pragma unroll
            for (int msk = 1; msk < 16; msk <<= 1)
                rs += __shfl_xor(rs, msk);
            lrow[j] = lrow[j] * alpha + rs;
            mrow[j] = mx;
            #pragma unroll
            for (int t = 0; t < 4; ++t) O[t][j] *= alpha;
        }

        // O += P * V  (K-dim = 64 keys, 2 chunks); Vt is [hd][key] so the
        // B fragment read is K-contiguous.
        #pragma unroll
        for (int kc = 0; kc < 64; kc += 32) {
            v16bf ap = frag_ld(&Pb[w][0] + kc, 64);
            #pragma unroll
            for (int t = 0; t < 4; ++t) {
                v16bf bv = frag_ld(Vt + (t * 16) * 64 + kc, 64);
                O[t] = wmma_bf16(ap, bv, O[t]);
            }
        }
    }

    // Normalize and write y (bf16) at [b, row, h*64 + col]
    #pragma unroll
    for (int j = 0; j < 8; ++j) {
        float inv = 1.0f / lrow[j];
        int row = qt * 64 + w * 16 + laneh * 8 + j;
        #pragma unroll
        for (int t = 0; t < 4; ++t)
            yb[(rowQ + row) * CEMB + hoff + t * 16 + lc] =
                (__bf16)(O[t][j] * inv);
    }
}

// ---------------------------------------------------------------------------
// Host launch
// ---------------------------------------------------------------------------
extern "C" void kernel_launch(void* const* d_in, const int* in_sizes, int n_in,
                              void* d_out, int out_size, void* d_ws, size_t ws_size,
                              hipStream_t stream) {
    const float* x      = (const float*)d_in[0];   // [4,2048,1024]
    const float* w_attn = (const float*)d_in[1];   // [3072,1024]
    const float* b_attn = (const float*)d_in[2];   // [3072]
    const float* w_proj = (const float*)d_in[3];   // [1024,1024]
    const float* b_proj = (const float*)d_in[4];   // [1024]
    float* out = (float*)d_out;                    // [4,2048,1024]

    char* ws = (char*)d_ws;
    __bf16* xb  = (__bf16*)(ws);                               // 16 MB
    __bf16* wab = (__bf16*)(ws + (size_t)16777216);            //  6 MB
    __bf16* wpb = (__bf16*)(ws + (size_t)23068672);            //  2 MB
    float*  qkv = (float* )(ws + (size_t)25165824);            // 96 MB
    __bf16* yb  = (__bf16*)(ws + (size_t)125829120);           // 16 MB

    // 1) bf16 conversions
    {
        int n4 = (MTOT * CEMB) / 4;          // 2,097,152
        cvt_f32_bf16<<<(n4 + 255) / 256, 256, 0, stream>>>(x, xb, n4);
        n4 = (C3 * CEMB) / 4;                //   786,432
        cvt_f32_bf16<<<(n4 + 255) / 256, 256, 0, stream>>>(w_attn, wab, n4);
        n4 = (CEMB * CEMB) / 4;              //   262,144
        cvt_f32_bf16<<<(n4 + 255) / 256, 256, 0, stream>>>(w_proj, wpb, n4);
    }

    // 2) qkv = x @ w_attn^T + b_attn   [8192 x 3072]
    gemm_bf16_abT<<<dim3(C3 / 128, MTOT / 128), 256, 0, stream>>>(
        xb, wab, b_attn, qkv, MTOT, C3, CEMB);

    // 3) flash attention -> yb (bf16)
    attn_flash<<<dim3(TSEQ / 64, NHEAD, BATCH), 128, 0, stream>>>(qkv, yb);

    // 4) out = y @ w_proj^T + b_proj   [8192 x 1024]
    gemm_bf16_abT<<<dim3(CEMB / 128, MTOT / 128), 256, 0, stream>>>(
        yb, wpb, b_proj, out, MTOT, CEMB, CEMB);
}
